// DecoderBlock_48868137894292
// MI455X (gfx1250) — matrix operand order repaired and checked by reading, not compiled, not run on hardware
//
#include <hip/hip_runtime.h>
#include <hip/hip_bf16.h>

// ---------------------------------------------------------------------------
// GPT decoder block on gfx1250 (MI455X): all GEMM-shaped work runs through
// v_wmma_f32_16x16x32_bf16 (bf16 inputs, fp32 accumulate).
// ---------------------------------------------------------------------------

#define EMB   1024
#define NSEQ  2048
#define NB    2
#define MROWS (NB * NSEQ)   // 4096 token rows

typedef __attribute__((ext_vector_type(16))) __bf16 v16bf;
typedef __attribute__((ext_vector_type(8)))  float  v8f;

union Frag { v16bf v; unsigned u[8]; };
struct U4 { unsigned x, y, z, w; };

__device__ __forceinline__ unsigned short f2bf(float f) {
  union { float f; unsigned u; } c; c.f = f;
  unsigned u = c.u;
  if ((u & 0x7f800000u) != 0x7f800000u)          // finite: round-to-nearest-even
    u += 0x7fffu + ((u >> 16) & 1u);
  return (unsigned short)(u >> 16);
}

__device__ __forceinline__ v8f zero8() {
  v8f z;
#pragma unroll
  for (int i = 0; i < 8; ++i) z[i] = 0.0f;
  return z;
}

__device__ __forceinline__ float gelu_tanh(float x) {
  const float c = 0.7978845608028654f;           // sqrt(2/pi)
  return 0.5f * x * (1.0f + tanhf(c * (x + 0.044715f * x * x * x)));
}

// ---------------------------------------------------------------------------
// fp32 -> bf16 weight cast
// ---------------------------------------------------------------------------
__global__ void cast_f32_bf16(const float* __restrict__ in,
                              unsigned short* __restrict__ out, int n) {
  int i = blockIdx.x * 256 + threadIdx.x;
  if (i < n) out[i] = f2bf(in[i]);
}

// ---------------------------------------------------------------------------
// LayerNorm over 1024 cols, writes bf16 (feeds the next GEMM's A operand)
// ---------------------------------------------------------------------------
__global__ __launch_bounds__(256) void layernorm_to_bf16(
    const float* __restrict__ X, const float* __restrict__ g,
    const float* __restrict__ b, unsigned short* __restrict__ Y) {
  __shared__ float red[256];
  const int row = blockIdx.x, tid = threadIdx.x;
  const float* x = X + (size_t)row * EMB;

  float s = 0.f;
  for (int c = tid; c < EMB; c += 256) s += x[c];
  red[tid] = s; __syncthreads();
  for (int o = 128; o > 0; o >>= 1) { if (tid < o) red[tid] += red[tid + o]; __syncthreads(); }
  float mean = red[0] * (1.0f / EMB);
  __syncthreads();

  float v = 0.f;
  for (int c = tid; c < EMB; c += 256) { float d = x[c] - mean; v += d * d; }
  red[tid] = v; __syncthreads();
  for (int o = 128; o > 0; o >>= 1) { if (tid < o) red[tid] += red[tid + o]; __syncthreads(); }
  float rstd = rsqrtf(red[0] * (1.0f / EMB) + 1e-5f);

  unsigned short* y = Y + (size_t)row * EMB;
  for (int c = tid; c < EMB; c += 256)
    y[c] = f2bf(g[c] * (x[c] - mean) * rstd + b[c]);
}

// ---------------------------------------------------------------------------
// bf16 WMMA GEMM: C[M,N] = A[M,K] * B[K,N]  (A,B bf16 row-major; fp32 acc)
// Block = 256 threads = 8 waves; tile 128x128x32; wave = 32x64 (2x4 WMMA tiles)
// EPI: 0 = fp32 store, 1 = +bias+resid -> fp32, 2 = +bias, GELU -> bf16
// ---------------------------------------------------------------------------
#define BM   128
#define BN   128
#define BK   32
#define LDT  34   // ushort LDS row stride (BK + 2, keeps dword alignment)
#define LDTD 17   // dword LDS row stride

template <int EPI>
__global__ __launch_bounds__(256, 2) void gemm_bf16_wmma(
    const unsigned short* __restrict__ A,   // [M][K] bf16 bits
    const unsigned short* __restrict__ Bw,  // [K][N] bf16 bits
    float* __restrict__ Cf, unsigned short* __restrict__ Cb,
    const float* __restrict__ bias, const float* __restrict__ resid,
    int M, int N, int K) {
  __shared__ __align__(16) unsigned short As[BM * LDT];
  __shared__ __align__(16) unsigned short Bs[BN * LDT];   // transposed: [n][k]
  unsigned* AsU = (unsigned*)As;
  unsigned* BsU = (unsigned*)Bs;

  const int tid = threadIdx.x;
  const int wave = tid >> 5, lane = tid & 31;
  const int l15 = lane & 15, lh = lane >> 4;
  const int wm = wave & 3, wn = wave >> 2;               // 4 x 2 wave grid
  const int bm0 = blockIdx.y * BM, bn0 = blockIdx.x * BN;

  v8f acc[2][4];
#pragma unroll
  for (int i = 0; i < 2; ++i)
#pragma unroll
    for (int j = 0; j < 4; ++j) acc[i][j] = zero8();

  for (int k0 = 0; k0 < K; k0 += BK) {
    // stage A tile (128x32 bf16), 16B vector loads
#pragma unroll
    for (int t = 0; t < 2; ++t) {
      int li = tid + t * 256;
      int row = li >> 2, ch = li & 3;
      U4 d = *(const U4*)(A + (size_t)(bm0 + row) * K + k0 + ch * 8);
      unsigned* dst = AsU + row * LDTD + ch * 4;
      dst[0] = d.x; dst[1] = d.y; dst[2] = d.z; dst[3] = d.w;
    }
    // stage B tile (32x128 bf16) transposed into [n][k]
#pragma unroll
    for (int t = 0; t < 2; ++t) {
      int li = tid + t * 256;
      int kr = li >> 4, ch = li & 15;
      U4 d = *(const U4*)(Bw + (size_t)(k0 + kr) * N + bn0 + ch * 8);
      unsigned uu[4] = { d.x, d.y, d.z, d.w };
#pragma unroll
      for (int j = 0; j < 4; ++j) {
        int n = ch * 8 + j * 2;
        Bs[n * LDT + kr]       = (unsigned short)(uu[j] & 0xffffu);
        Bs[(n + 1) * LDT + kr] = (unsigned short)(uu[j] >> 16);
      }
    }
    __syncthreads();

    // fragment assembly, contiguous k order (k = 16*lh + s) in both A and B
    Frag a[2], b[4];
#pragma unroll
    for (int am = 0; am < 2; ++am) {
      int rowA = wm * 32 + am * 16 + l15;
#pragma unroll
      for (int v = 0; v < 8; ++v) {
        int dv = lh * 8 + v;                              // K pair index (dwords)
        a[am].u[v] = AsU[rowA * LDTD + dv];
      }
    }
#pragma unroll
    for (int bn = 0; bn < 4; ++bn) {
      int rowB = wn * 64 + bn * 16 + l15;
#pragma unroll
      for (int v = 0; v < 8; ++v)
        b[bn].u[v] = BsU[rowB * LDTD + lh * 8 + v];      // 16 contiguous K per lane
    }
#pragma unroll
    for (int am = 0; am < 2; ++am)
#pragma unroll
      for (int bn = 0; bn < 4; ++bn)
        acc[am][bn] = __builtin_amdgcn_wmma_f32_16x16x32_bf16(
            false, a[am].v, false, b[bn].v, (short)0, acc[am][bn], false, false);
    __syncthreads();
  }

  // epilogue (C layout: lane n = lane&15, row = vgpr + 8*(lane>>4))
#pragma unroll
  for (int am = 0; am < 2; ++am)
#pragma unroll
    for (int bn = 0; bn < 4; ++bn)
#pragma unroll
      for (int r = 0; r < 8; ++r) {
        int m = bm0 + wm * 32 + am * 16 + lh * 8 + r;
        int n = bn0 + wn * 64 + bn * 16 + l15;
        float val = acc[am][bn][r];
        size_t idx = (size_t)m * N + n;
        if (EPI == 0) {
          Cf[idx] = val;
        } else if (EPI == 1) {
          Cf[idx] = val + bias[n] + resid[idx];
        } else {
          Cb[idx] = f2bf(gelu_tanh(val + bias[n]));
        }
      }
}

// ---------------------------------------------------------------------------
// Causal flash attention, one block = (batch, head, 64-query tile), 4 waves.
// QK^T and PV through bf16 WMMA; O accumulator lives in registers and the
// softmax rescale is folded into the WMMA C operand (D = P*V + corr*O).
// ---------------------------------------------------------------------------
#define LSD 66   // ushort stride for Qs/Ks/Vs, float stride for SP

__global__ __launch_bounds__(128, 2) void attn_flash_wmma(
    const float* __restrict__ qkv, unsigned short* __restrict__ ctx) {
  __shared__ __align__(16) unsigned short Qs[64 * LSD];
  __shared__ __align__(16) unsigned short Ks[64 * LSD];
  __shared__ __align__(16) unsigned short Vs[64 * LSD];  // transposed: [d][key]
  __shared__ __align__(16) float SP[64 * LSD];           // S(f32) overlaid by P(bf16)
  __shared__ float mi[64], li[64], cf[64];               // running max/sum, rescale

  const int tid = threadIdx.x;
  const int wave = tid >> 5, lane = tid & 31;
  const int l15 = lane & 15, lh = lane >> 4;
  const int qt = blockIdx.x, h = blockIdx.y, bb = blockIdx.z;
  const size_t rowbase = (size_t)bb * NSEQ;
  const int qcol = h * 64, kcol = EMB + h * 64, vcol = 2 * EMB + h * 64;
  const float NEG = -__builtin_inff();

  for (int i = tid; i < 64 * 64; i += 128) {
    int r = i >> 6, d = i & 63;
    float qv = qkv[(rowbase + qt * 64 + r) * (3 * EMB) + qcol + d];
    Qs[r * LSD + d] = f2bf(qv * 0.125f);   // 1/sqrt(64) folded into Q
  }
  if (tid < 64) { mi[tid] = NEG; li[tid] = 0.f; }
  __syncthreads();

  const unsigned* QsU = (const unsigned*)Qs;
  const unsigned* KsU = (const unsigned*)Ks;
  const unsigned* VsU = (const unsigned*)Vs;
  const unsigned* PsU = (const unsigned*)SP;
  const int mrow = wave * 16 + l15;       // A-fragment row for this lane
  const int orow = wave * 16 + lh * 8;    // first C-row owned by this lane

  v8f oacc[4];                            // O: 16 q-rows x 64 d per wave
#pragma unroll
  for (int dt = 0; dt < 4; ++dt) oacc[dt] = zero8();

  for (int kt = 0; kt <= qt; ++kt) {
    for (int i = tid; i < 64 * 64; i += 128) {
      int r = i >> 6, d = i & 63;
      size_t base = (rowbase + kt * 64 + r) * (3 * EMB);
      Ks[r * LSD + d] = f2bf(qkv[base + kcol + d]);
      Vs[d * LSD + r] = f2bf(qkv[base + vcol + d]);
    }
    __syncthreads();

    // S = Q K^T : each wave -> 16 q-rows x 64 keys (4 tiles x 2 k-steps)
#pragma unroll
    for (int nt = 0; nt < 4; ++nt) {
      v8f s = zero8();
#pragma unroll
      for (int ks = 0; ks < 2; ++ks) {
        Frag a, b;
#pragma unroll
        for (int v = 0; v < 8; ++v) {
          int dv = lh * 8 + ks * 16 + v;
          a.u[v] = QsU[mrow * (LSD / 2) + dv];
          b.u[v] = KsU[(nt * 16 + l15) * (LSD / 2) + lh * 8 + ks * 16 + v];
        }
        s = __builtin_amdgcn_wmma_f32_16x16x32_bf16(false, a.v, false, b.v,
                                                    (short)0, s, false, false);
      }
#pragma unroll
      for (int r = 0; r < 8; ++r) {
        int m = orow + r;
        int n = nt * 16 + l15;
        int gq = qt * 64 + m, gk = kt * 64 + n;
        SP[m * LSD + n] = (gk <= gq) ? s[r] : NEG;       // causal mask
      }
    }
    __syncthreads();

    // online softmax: thread q owns row q; publishes rescale factor cf[q]
    if (tid < 64) {
      int q = tid;
      float mold = mi[q], rowmax = NEG;
      for (int k2 = 0; k2 < 64; ++k2) rowmax = fmaxf(rowmax, SP[q * LSD + k2]);
      float mnew = fmaxf(mold, rowmax);
      float corr = (mold == NEG) ? 0.f : __expf(mold - mnew);
      unsigned short* P = (unsigned short*)SP;
      float lsum = 0.f;
      for (int k2 = 0; k2 < 64; ++k2) {
        float p = __expf(SP[q * LSD + k2] - mnew);       // exp(-inf)=0 on mask
        lsum += p;
        P[q * (2 * LSD) + k2] = f2bf(p);   // in-place overlay: 2B write trails 4B read
      }
      cf[q] = corr;
      mi[q] = mnew;
      li[q] = li[q] * corr + lsum;
    }
    __syncthreads();

    // O = P V + corr*O : rescale folded into the WMMA C operand
    float cr[8];
#pragma unroll
    for (int r = 0; r < 8; ++r) cr[r] = cf[orow + r];
#pragma unroll
    for (int dt = 0; dt < 4; ++dt) {
      v8f c;
#pragma unroll
      for (int r = 0; r < 8; ++r) c[r] = oacc[dt][r] * cr[r];
#pragma unroll
      for (int ks = 0; ks < 2; ++ks) {
        Frag a, b;
#pragma unroll
        for (int v = 0; v < 8; ++v) {
          int dv = lh * 8 + ks * 16 + v;
          a.u[v] = PsU[mrow * LSD + dv];                 // P row = 2*LSD ushorts = LSD dwords
          b.u[v] = VsU[(dt * 16 + l15) * (LSD / 2) + lh * 8 + ks * 16 + v];
        }
        c = __builtin_amdgcn_wmma_f32_16x16x32_bf16(false, a.v, false, b.v,
                                                    (short)0, c, false, false);
      }
      oacc[dt] = c;
    }
    __syncthreads();
  }

  // finalize: divide by running sum, write bf16 ctx straight from registers
  float inv[8];
#pragma unroll
  for (int r = 0; r < 8; ++r) inv[r] = 1.0f / li[orow + r];
#pragma unroll
  for (int dt = 0; dt < 4; ++dt)
#pragma unroll
    for (int r = 0; r < 8; ++r) {
      int m = orow + r;
      int n = dt * 16 + l15;
      ctx[(rowbase + qt * 64 + m) * EMB + h * 64 + n] = f2bf(oacc[dt][r] * inv[r]);
    }
}

// ---------------------------------------------------------------------------
// Host-side orchestration
// ---------------------------------------------------------------------------
extern "C" void kernel_launch(void* const* d_in, const int* in_sizes, int n_in,
                              void* d_out, int out_size, void* d_ws, size_t ws_size,
                              hipStream_t stream) {
  const float* x      = (const float*)d_in[0];
  const float* ln1_w  = (const float*)d_in[1];
  const float* ln1_b  = (const float*)d_in[2];
  const float* w_qkv  = (const float*)d_in[3];
  const float* w_attn = (const float*)d_in[4];
  const float* b_attn = (const float*)d_in[5];
  const float* ln2_w  = (const float*)d_in[6];
  const float* ln2_b  = (const float*)d_in[7];
  const float* w_fc   = (const float*)d_in[8];
  const float* b_fc   = (const float*)d_in[9];
  const float* w_proj = (const float*)d_in[10];
  const float* b_proj = (const float*)d_in[11];
  float* out = (float*)d_out;

  char* ws = (char*)d_ws;
  size_t off = 0;
  auto carve = [&](size_t bytes) {
    char* p = ws + off;
    off += (bytes + 255) & ~(size_t)255;
    return p;
  };

  unsigned short* h1   = (unsigned short*)carve((size_t)MROWS * EMB * 2);
  unsigned short* wqkv = (unsigned short*)carve((size_t)EMB * 3 * EMB * 2);
  unsigned short* wat  = (unsigned short*)carve((size_t)EMB * EMB * 2);
  unsigned short* wfc  = (unsigned short*)carve((size_t)EMB * 4 * EMB * 2);
  unsigned short* wpr  = (unsigned short*)carve((size_t)4 * EMB * EMB * 2);
  float*          qkv  = (float*)carve((size_t)MROWS * 3 * EMB * 4);
  unsigned short* ctx  = (unsigned short*)carve((size_t)MROWS * EMB * 2);
  float*          x1   = (float*)carve((size_t)MROWS * EMB * 4);
  unsigned short* h2   = (unsigned short*)carve((size_t)MROWS * EMB * 2);
  unsigned short* fco  = (unsigned short*)carve((size_t)MROWS * 4 * EMB * 2);

  int n;
  n = EMB * 3 * EMB; cast_f32_bf16<<<(n + 255) / 256, 256, 0, stream>>>(w_qkv, wqkv, n);
  n = EMB * EMB;     cast_f32_bf16<<<(n + 255) / 256, 256, 0, stream>>>(w_attn, wat, n);
  n = EMB * 4 * EMB; cast_f32_bf16<<<(n + 255) / 256, 256, 0, stream>>>(w_fc, wfc, n);
  n = 4 * EMB * EMB; cast_f32_bf16<<<(n + 255) / 256, 256, 0, stream>>>(w_proj, wpr, n);

  // h1 = bf16(LN1(x))
  layernorm_to_bf16<<<MROWS, 256, 0, stream>>>(x, ln1_w, ln1_b, h1);

  // qkv = h1 @ w_qkv  (fp32 out)
  gemm_bf16_wmma<0><<<dim3(3 * EMB / BN, MROWS / BM), 256, 0, stream>>>(
      h1, wqkv, qkv, nullptr, nullptr, nullptr, MROWS, 3 * EMB, EMB);

  // ctx = causal softmax attention (bf16 out)
  attn_flash_wmma<<<dim3(NSEQ / 64, 16, NB), 128, 0, stream>>>(qkv, ctx);

  // x1 = x + ctx @ w_attn + b_attn
  gemm_bf16_wmma<1><<<dim3(EMB / BN, MROWS / BM), 256, 0, stream>>>(
      ctx, wat, x1, nullptr, b_attn, x, MROWS, EMB, EMB);

  // h2 = bf16(LN2(x1))
  layernorm_to_bf16<<<MROWS, 256, 0, stream>>>(x1, ln2_w, ln2_b, h2);

  // fco = bf16(gelu(h2 @ w_fc + b_fc))
  gemm_bf16_wmma<2><<<dim3(4 * EMB / BN, MROWS / BM), 256, 0, stream>>>(
      h2, wfc, nullptr, fco, b_fc, nullptr, MROWS, 4 * EMB, EMB);

  // out = x1 + fco @ w_proj + b_proj
  gemm_bf16_wmma<1><<<dim3(EMB / BN, MROWS / BM), 256, 0, stream>>>(
      fco, wpr, out, nullptr, b_proj, x1, MROWS, EMB, 4 * EMB);
}
